// HierarchicalLSTM_1133871366708
// MI455X (gfx1250) — matrix operand order repaired and checked by reading, not compiled, not run on hardware
//
#include <hip/hip_runtime.h>

// ---------------------------------------------------------------------------
// Problem constants (match reference)
// ---------------------------------------------------------------------------
#define S_   64
#define B_   32
#define L_   64
#define V_   32000
#define D_   300
#define DP_  320          // D padded to multiple of 32 (WMMA K granule)
#define HH_  256          // message-LSTM hidden (per direction)
#define H_   512          // conv-LSTM hidden
#define F_   16
#define NSEQ_ (S_*B_)     // 2048 messages
#define G1_  (4*HH_)      // 1024 message gates
#define G2_  (4*H_)       // 2048 conv gates

typedef __attribute__((ext_vector_type(16))) __bf16 v16bf;
typedef __attribute__((ext_vector_type(8)))  float  v8f;

__device__ inline float sigmoidf_(float x) { return 1.f / (1.f + __expf(-x)); }

// ---------------------------------------------------------------------------
// WMMA fragment loaders (CDNA5 ISA 7.12.2, wave32)
// A (16-bit, 16x32 MxK): lane row=lane&15; lanes<16 hold K=[0..7]+[16..23],
//                        lanes>=16 hold K=[8..15]+[24..31]  -> two b128 loads
// B (32x16 KxN) from row-major W[gate][k]: lane col=lane&15; lanes<16 K=[0..7]+[16..23],
//                        lanes>=16 K=[8..15]+[24..31] -> two b128 loads
// C/D (f32 16x16): lane n=lane&15, VGPR r -> m = r + (lane<16 ? 0 : 8)
// ---------------------------------------------------------------------------
__device__ inline v16bf load_a16(const __bf16* base, int ld, int k0, int lane) {
  const int row = lane & 15;
  const int kb  = (lane < 16) ? 0 : 8;
  const __bf16* p = base + row * ld + k0 + kb;
  union { v16bf v; uint4 q[2]; } u;
  u.q[0] = *(const uint4*)(p);        // halves 0..7  : K = kb+0..7
  u.q[1] = *(const uint4*)(p + 16);   // halves 8..15 : K = kb+16..23
  return u.v;
}

__device__ inline v16bf load_b16(const __bf16* w, int n0, int ld, int k0, int lane) {
  const int c  = lane & 15;
  const int kb = (lane < 16) ? 0 : 8;
  const __bf16* p = w + (size_t)(n0 + c) * ld + k0 + kb;
  union { v16bf v; uint4 q[2]; } u;
  u.q[0] = *(const uint4*)(p);
  u.q[1] = *(const uint4*)(p + 16);
  return u.v;
}

__device__ inline v8f wmma_bf16(v16bf a, v16bf b, v8f c) {
  return __builtin_amdgcn_wmma_f32_16x16x32_bf16(false, a, false, b, (short)0, c,
                                                 false, false);
}

// ---------------------------------------------------------------------------
// Kernel 0: weights + embedding table -> bf16 (Wih_f/b padded 300->320)
// ---------------------------------------------------------------------------
__global__ void cvt_weights_kernel(
    const float* __restrict__ E,
    const float* __restrict__ wihF, const float* __restrict__ whhF,
    const float* __restrict__ wihB, const float* __restrict__ whhB,
    const float* __restrict__ wihC, const float* __restrict__ whhC,
    __bf16* __restrict__ o_E,
    __bf16* __restrict__ o_wihF, __bf16* __restrict__ o_whhF,
    __bf16* __restrict__ o_wihB, __bf16* __restrict__ o_whhB,
    __bf16* __restrict__ o_wihC, __bf16* __restrict__ o_whhC) {
  const int stride = gridDim.x * blockDim.x;
  const int t0 = blockIdx.x * blockDim.x + threadIdx.x;
  for (size_t i = t0; i < (size_t)V_ * D_; i += stride)
    o_E[i] = (__bf16)E[i];
  for (int i = t0; i < G1_ * DP_; i += stride) {
    int r = i / DP_, k = i - r * DP_;
    float vf = (k < D_) ? wihF[(size_t)r * D_ + k] : 0.f;
    float vb = (k < D_) ? wihB[(size_t)r * D_ + k] : 0.f;
    o_wihF[i] = (__bf16)vf;
    o_wihB[i] = (__bf16)vb;
  }
  for (int i = t0; i < G1_ * HH_; i += stride) {
    o_whhF[i] = (__bf16)whhF[i];
    o_whhB[i] = (__bf16)whhB[i];
  }
  for (int i = t0; i < G2_ * H_; i += stride) {
    o_wihC[i] = (__bf16)wihC[i];
    o_whhC[i] = (__bf16)whhC[i];
  }
}

// ---------------------------------------------------------------------------
// Kernel 1: message-level LSTM (one direction per blockIdx.y), 16 seqs/block.
// 8 waves; wave w owns hidden units [32w,32w+32) for all 4 gates ->
// 8 WMMA accumulators/wave, elementwise LSTM fully wave-local.
// Produces enc (time-max of masked h), bf16, fwd||bwd concatenated.
// ---------------------------------------------------------------------------
__global__ __launch_bounds__(256) void msg_lstm_kernel(
    const int*   __restrict__ messages, const __bf16* __restrict__ E_b,
    const __bf16* __restrict__ wihF, const __bf16* __restrict__ whhF,
    const float* __restrict__ bF,
    const __bf16* __restrict__ wihB, const __bf16* __restrict__ whhB,
    const float* __restrict__ bB,
    __bf16* __restrict__ enc_b) {
  __shared__ __bf16 xs[16 * DP_];    // x_t tile, bf16, padded K
  __shared__ __bf16 hsm[16 * HH_];   // h state tile, bf16
  __shared__ int mlen[16];

  const int tid  = threadIdx.x;
  const int lane = tid & 31;
  const int w    = tid >> 5;
  const int col  = lane & 15;
  const int hi   = (lane < 16) ? 0 : 8;
  const int jbase = w * 32;
  const int n0   = blockIdx.x * 16;
  const int dir  = blockIdx.y;

  const __bf16* Wih  = dir ? wihB : wihF;
  const __bf16* Whh  = dir ? whhB : whhF;
  const float*  bias = dir ? bB : bF;

  if (tid < 16) {
    const int* mrow = messages + (size_t)(n0 + tid) * L_;
    int cnt = 0;
    for (int l = 0; l < L_; ++l) cnt += (mrow[l] != 0) ? 1 : 0;
    mlen[tid] = cnt > 0 ? cnt : 1;
  }
  for (int e = tid; e < 16 * HH_; e += 256) hsm[e] = (__bf16)0.f;
  __syncthreads();

  int ml[8];
#pragma unroll
  for (int r = 0; r < 8; ++r) ml[r] = mlen[r + hi];

  v8f cst[2], hreg[2], hm[2];
#pragma unroll
  for (int sb = 0; sb < 2; ++sb)
#pragma unroll
    for (int r = 0; r < 8; ++r) {
      cst[sb][r] = 0.f; hreg[sb][r] = 0.f; hm[sb][r] = -1e30f;
    }

  for (int t = 0; t < L_; ++t) {
    const int l = dir ? (L_ - 1 - t) : t;

    // gather embedded x_t into LDS: 8-byte (4 x bf16) chunks straight from
    // the bf16 embedding table; zero-pad cols [300,320).
    // 16 rows x 80 chunks = 1280 chunks, 5 per thread.
    for (int e = tid; e < 16 * (DP_ / 4); e += 256) {
      const int m  = e / (DP_ / 4);
      const int k  = (e - m * (DP_ / 4)) * 4;
      uint2 val;
      if (k < D_) {
        int token = messages[(size_t)(n0 + m) * L_ + l];
        val = *(const uint2*)(E_b + (size_t)token * D_ + k);
      } else {
        val.x = 0u; val.y = 0u;
      }
      *(uint2*)(xs + m * DP_ + k) = val;
    }
    __syncthreads();

    v8f acc[4][2];
#pragma unroll
    for (int g = 0; g < 4; ++g)
#pragma unroll
      for (int sb = 0; sb < 2; ++sb) {
        float bv = bias[g * HH_ + jbase + sb * 16 + col];
#pragma unroll
        for (int r = 0; r < 8; ++r) acc[g][sb][r] = bv;
      }

    // x_t @ Wih^T   (K = 320)
#pragma unroll 2
    for (int kt = 0; kt < DP_ / 32; ++kt) {
      v16bf a = load_a16(xs, DP_, kt * 32, lane);
#pragma unroll
      for (int g = 0; g < 4; ++g)
#pragma unroll
        for (int sb = 0; sb < 2; ++sb) {
          v16bf bm = load_b16(Wih, g * HH_ + jbase + sb * 16, DP_, kt * 32, lane);
          acc[g][sb] = wmma_bf16(a, bm, acc[g][sb]);
        }
    }
    // h_{t-1} @ Whh^T (K = 256)
#pragma unroll 2
    for (int kt = 0; kt < HH_ / 32; ++kt) {
      v16bf a = load_a16(hsm, HH_, kt * 32, lane);
#pragma unroll
      for (int g = 0; g < 4; ++g)
#pragma unroll
        for (int sb = 0; sb < 2; ++sb) {
          v16bf bm = load_b16(Whh, g * HH_ + jbase + sb * 16, HH_, kt * 32, lane);
          acc[g][sb] = wmma_bf16(a, bm, acc[g][sb]);
        }
    }
    __syncthreads();

    // elementwise LSTM + masking + running max-pool (all wave-local)
#pragma unroll
    for (int sb = 0; sb < 2; ++sb)
#pragma unroll
      for (int r = 0; r < 8; ++r) {
        float iv = sigmoidf_(acc[0][sb][r]);
        float fv = sigmoidf_(acc[1][sb][r]);
        float gv = tanhf(acc[2][sb][r]);
        float ov = sigmoidf_(acc[3][sb][r]);
        float cn = fv * cst[sb][r] + iv * gv;
        float hn = ov * tanhf(cn);
        bool valid = (l < ml[r]);
        cst[sb][r]  = valid ? cn : cst[sb][r];
        hreg[sb][r] = valid ? hn : hreg[sb][r];
        float hv = valid ? hn : 0.f;          // pad_packed: invalid steps -> 0
        hm[sb][r] = fmaxf(hm[sb][r], hv);
        hsm[(r + hi) * HH_ + jbase + sb * 16 + col] = (__bf16)hreg[sb][r];
      }
    // next iteration's x-gather + sync fences hsm writes vs. reads
  }

  // enc[n, dir*256 + j] = max_t masked h
#pragma unroll
  for (int sb = 0; sb < 2; ++sb)
#pragma unroll
    for (int r = 0; r < 8; ++r)
      enc_b[(size_t)(n0 + r + hi) * H_ + dir * HH_ + jbase + sb * 16 + col] =
          (__bf16)hm[sb][r];
}

// ---------------------------------------------------------------------------
// Kernel 2: xg_c = enc @ Wih_c^T + b_c   (2048x512 x 512x2048, f32 out)
// ---------------------------------------------------------------------------
__global__ __launch_bounds__(256) void xgc_gemm_kernel(
    const __bf16* __restrict__ enc_b, const __bf16* __restrict__ wihC,
    const float* __restrict__ bC, float* __restrict__ xgc) {
  const int tid = threadIdx.x, lane = tid & 31, w = tid >> 5;
  const int col = lane & 15, hi = (lane < 16) ? 0 : 8;
  const int row0 = blockIdx.x * 16;
  const int ncol = blockIdx.y * 128 + w * 16;

  v8f acc;
  float bv = bC[ncol + col];
#pragma unroll
  for (int r = 0; r < 8; ++r) acc[r] = bv;

#pragma unroll 4
  for (int kt = 0; kt < H_ / 32; ++kt) {
    v16bf a = load_a16(enc_b + (size_t)row0 * H_, H_, kt * 32, lane);
    v16bf b = load_b16(wihC, ncol, H_, kt * 32, lane);
    acc = wmma_bf16(a, b, acc);
  }
#pragma unroll
  for (int r = 0; r < 8; ++r)
    xgc[(size_t)(row0 + r + hi) * G2_ + ncol + col] = acc[r];
}

// ---------------------------------------------------------------------------
// Kernel 3: conversation-level LSTM, single block (serial in S, B=32 batch).
// 16 waves; wave w owns hidden units [32w,32w+32) for all gates, 2 M-tiles.
// ---------------------------------------------------------------------------
__global__ __launch_bounds__(512) void conv_lstm_kernel(
    const float* __restrict__ xgc, const __bf16* __restrict__ whhC,
    const int* __restrict__ conv_len, float* __restrict__ convh) {
  __shared__ __bf16 hs2[32 * H_];   // 32 KB
  __shared__ int clen[32];

  const int tid = threadIdx.x, lane = tid & 31, w = tid >> 5;
  const int col = lane & 15, hi = (lane < 16) ? 0 : 8;
  const int jbase = w * 32;

  if (tid < 32) clen[tid] = conv_len[tid];
  for (int e = tid; e < 32 * H_; e += 512) hs2[e] = (__bf16)0.f;
  __syncthreads();

  int cl[2][8];
#pragma unroll
  for (int mt = 0; mt < 2; ++mt)
#pragma unroll
    for (int r = 0; r < 8; ++r) cl[mt][r] = clen[mt * 16 + r + hi];

  v8f cst[2][2], hreg[2][2];   // [mt][sb]
#pragma unroll
  for (int mt = 0; mt < 2; ++mt)
#pragma unroll
    for (int sb = 0; sb < 2; ++sb)
#pragma unroll
      for (int r = 0; r < 8; ++r) { cst[mt][sb][r] = 0.f; hreg[mt][sb][r] = 0.f; }

  for (int s = 0; s < S_; ++s) {
    v8f acc[4][2][2];  // [gate][sb][mt]
#pragma unroll
    for (int g = 0; g < 4; ++g)
#pragma unroll
      for (int sb = 0; sb < 2; ++sb)
#pragma unroll
        for (int mt = 0; mt < 2; ++mt)
#pragma unroll
          for (int r = 0; r < 8; ++r)
            acc[g][sb][mt][r] =
                xgc[(size_t)(s * B_ + mt * 16 + r + hi) * G2_ +
                    g * H_ + jbase + sb * 16 + col];

    // + h_{s-1} @ Whh_c^T  (K = 512)
#pragma unroll 1
    for (int kt = 0; kt < H_ / 32; ++kt) {
      v16bf a0 = load_a16(hs2, H_, kt * 32, lane);
      v16bf a1 = load_a16(hs2 + 16 * H_, H_, kt * 32, lane);
#pragma unroll
      for (int g = 0; g < 4; ++g)
#pragma unroll
        for (int sb = 0; sb < 2; ++sb) {
          v16bf bm = load_b16(whhC, g * H_ + jbase + sb * 16, H_, kt * 32, lane);
          acc[g][sb][0] = wmma_bf16(a0, bm, acc[g][sb][0]);
          acc[g][sb][1] = wmma_bf16(a1, bm, acc[g][sb][1]);
        }
    }
    __syncthreads();

#pragma unroll
    for (int mt = 0; mt < 2; ++mt)
#pragma unroll
      for (int sb = 0; sb < 2; ++sb)
#pragma unroll
        for (int r = 0; r < 8; ++r) {
          int m = mt * 16 + r + hi;
          float iv = sigmoidf_(acc[0][sb][mt][r]);
          float fv = sigmoidf_(acc[1][sb][mt][r]);
          float gv = tanhf(acc[2][sb][mt][r]);
          float ov = sigmoidf_(acc[3][sb][mt][r]);
          float cn = fv * cst[mt][sb][r] + iv * gv;
          float hn = ov * tanhf(cn);
          bool valid = (s < cl[mt][r]);
          cst[mt][sb][r]  = valid ? cn : cst[mt][sb][r];
          hreg[mt][sb][r] = valid ? hn : hreg[mt][sb][r];
          hs2[m * H_ + jbase + sb * 16 + col] = (__bf16)hreg[mt][sb][r];
          convh[(size_t)(s * B_ + m) * H_ + jbase + sb * 16 + col] =
              valid ? hn : 0.f;
        }
    __syncthreads();
  }
}

// ---------------------------------------------------------------------------
// Kernel 4: logits = [conv_out, game_features] @ Wc + bc
// ---------------------------------------------------------------------------
__global__ void final_proj_kernel(const float* __restrict__ convh,
                                  const float* __restrict__ gf,
                                  const float* __restrict__ Wc,
                                  const float* __restrict__ bc,
                                  float* __restrict__ out) {
  int n = blockIdx.x * blockDim.x + threadIdx.x;
  if (n >= NSEQ_) return;
  float s = bc[0];
  const float* hrow = convh + (size_t)n * H_;
  for (int j = 0; j < H_; ++j) s += hrow[j] * Wc[j];
  const float* grow = gf + (size_t)n * F_;
  for (int f = 0; f < F_; ++f) s += grow[f] * Wc[H_ + f];
  out[n] = s;
}

// ---------------------------------------------------------------------------
// Host launcher
// ---------------------------------------------------------------------------
extern "C" void kernel_launch(void* const* d_in, const int* in_sizes, int n_in,
                              void* d_out, int out_size, void* d_ws, size_t ws_size,
                              hipStream_t stream) {
  const int*   messages = (const int*)d_in[0];
  const float* gamef    = (const float*)d_in[1];
  const int*   convlen  = (const int*)d_in[2];
  const float* E        = (const float*)d_in[3];
  const float* wihF     = (const float*)d_in[4];
  const float* whhF     = (const float*)d_in[5];
  const float* bF       = (const float*)d_in[6];
  const float* wihB     = (const float*)d_in[7];
  const float* whhB     = (const float*)d_in[8];
  const float* bB       = (const float*)d_in[9];
  const float* wihC     = (const float*)d_in[10];
  const float* whhC     = (const float*)d_in[11];
  const float* bC       = (const float*)d_in[12];
  const float* Wc       = (const float*)d_in[13];
  const float* bc       = (const float*)d_in[14];
  float* out = (float*)d_out;

  char* p = (char*)d_ws;
  auto take = [&](size_t bytes) -> char* {
    char* q = p;
    p += (bytes + 255) & ~(size_t)255;
    return q;
  };
  __bf16* E_b    = (__bf16*)take((size_t)V_ * D_ * 2);     // 19.2 MB
  __bf16* wihF_b = (__bf16*)take((size_t)G1_ * DP_ * 2);
  __bf16* wihB_b = (__bf16*)take((size_t)G1_ * DP_ * 2);
  __bf16* whhF_b = (__bf16*)take((size_t)G1_ * HH_ * 2);
  __bf16* whhB_b = (__bf16*)take((size_t)G1_ * HH_ * 2);
  __bf16* wihC_b = (__bf16*)take((size_t)G2_ * H_ * 2);
  __bf16* whhC_b = (__bf16*)take((size_t)G2_ * H_ * 2);
  __bf16* enc_b  = (__bf16*)take((size_t)NSEQ_ * H_ * 2);
  float*  xgc    = (float*)take((size_t)NSEQ_ * G2_ * 4);
  float*  convh  = (float*)take((size_t)NSEQ_ * H_ * 4);

  cvt_weights_kernel<<<2048, 256, 0, stream>>>(
      E, wihF, whhF, wihB, whhB, wihC, whhC,
      E_b, wihF_b, whhF_b, wihB_b, whhB_b, wihC_b, whhC_b);

  msg_lstm_kernel<<<dim3(NSEQ_ / 16, 2), 256, 0, stream>>>(
      messages, E_b, wihF_b, whhF_b, bF, wihB_b, whhB_b, bB, enc_b);

  xgc_gemm_kernel<<<dim3(NSEQ_ / 16, G2_ / 128), 256, 0, stream>>>(
      enc_b, wihC_b, bC, xgc);

  conv_lstm_kernel<<<1, 512, 0, stream>>>(xgc, whhC_b, convlen, convh);

  final_proj_kernel<<<NSEQ_ / 256, 256, 0, stream>>>(convh, gamef, Wc, bc, out);
}
